// GNNLayer_20512763806083
// MI455X (gfx1250) — matrix operand order repaired and checked by reading, not compiled, not run on hardware
//
#include <hip/hip_runtime.h>

#define BB 8
#define VV 256
#define HH 128
#define BV (BB * VV)                 // 2048
#define BVH (BV * HH)                // 262144
#define NE_ELEMS (67108864ull)       // 8*256*256*128

typedef float  v8f   __attribute__((ext_vector_type(8)));
typedef __bf16 v16bf __attribute__((ext_vector_type(16)));

// fast sigmoid: v_exp_f32 + v_rcp_f32 (avoids IEEE div expansion)
__device__ __forceinline__ float fast_sigmoid(float x) {
  return __builtin_amdgcn_rcpf(1.0f + __expf(-x));
}

// ---------------------------------------------------------------------------
// K0a: convert C_w (128x128 f32, row-major [o][k]) to bf16; zero stat accums.
// stats layout (floats): [0..127]=esum [128..255]=esumsq [256..383]=hsum [384..511]=hsumsq
// ---------------------------------------------------------------------------
__global__ void gcn_init(const float* __restrict__ Cw, __bf16* __restrict__ cwb,
                         float* __restrict__ stats) {
  int gid = blockIdx.x * blockDim.x + threadIdx.x;
  if (gid < HH * HH) cwb[gid] = (__bf16)Cw[gid];
  if (gid < 512) stats[gid] = 0.0f;
}

// ---------------------------------------------------------------------------
// K0b: the four H x H linears on h (tiny: 268 MFLOP).
//   Uh = h@Uw^T + Ub ; Vh = h@Vw^T + Vb ; Ahat = h@Aw^T + Ab + Cb ; Bh = h@Bw^T + Bb
// ---------------------------------------------------------------------------
__global__ void gcn_linear4(const float* __restrict__ h,
                            const float* __restrict__ Uw, const float* __restrict__ Ub,
                            const float* __restrict__ Vw, const float* __restrict__ Vb,
                            const float* __restrict__ Aw, const float* __restrict__ Ab,
                            const float* __restrict__ Bw, const float* __restrict__ Bb,
                            const float* __restrict__ Cb,
                            float* __restrict__ Uh, float* __restrict__ Vh,
                            float* __restrict__ Ahat, float* __restrict__ Bh) {
  int gid = blockIdx.x * blockDim.x + threadIdx.x;  // 0..BVH-1
  int row = gid >> 7, o = gid & 127;
  const float* hr = h + (size_t)row * HH;
  const float* uw = Uw + (size_t)o * HH;
  const float* vw = Vw + (size_t)o * HH;
  const float* aw = Aw + (size_t)o * HH;
  const float* bw = Bw + (size_t)o * HH;
  float aU = 0.f, aV = 0.f, aA = 0.f, aB = 0.f;
#pragma unroll 8
  for (int k = 0; k < HH; ++k) {
    float hv = hr[k];
    aU = fmaf(hv, uw[k], aU);
    aV = fmaf(hv, vw[k], aV);
    aA = fmaf(hv, aw[k], aA);
    aB = fmaf(hv, bw[k], aB);
  }
  Uh[gid]   = aU + Ub[o];
  Vh[gid]   = aV + Vb[o];
  Ahat[gid] = aA + Ab[o] + Cb[o];   // fold C bias into Ah broadcast term
  Bh[gid]   = aB + Bb[o];
}

// ---------------------------------------------------------------------------
// K1: main fused kernel. One 256-thread WG (8 waves) per (b,i).
//   GEMM: Ce[j,o] = sum_k e[b,i,j,k] * Cw[o,k]  via v_wmma_f32_16x16x32_bf16
//   e_pre[b,i,j,o] = Ce + Ahat[b,j,o] + Bh[b,i,o]   -> written to out_e (pre-BN)
//   agg[b,i,o]    += graph[b,i,j] ? 0 : sigmoid(e_pre)*Vh[b,j,o]
//   h_pre[b,i,o]   = Uh[b,i,o] + agg                -> written to out_h (pre-BN)
//   per-channel BN moments accumulated to global stats via atomics.
// ---------------------------------------------------------------------------
__global__ void __launch_bounds__(256)
gcn_main(const float* __restrict__ e, const unsigned char* __restrict__ graph,
         const float* __restrict__ Ahat, const float* __restrict__ Vh,
         const float* __restrict__ Bh, const float* __restrict__ Uh,
         const __bf16* __restrict__ cwb,
         float* __restrict__ out_e, float* __restrict__ out_h,
         float* __restrict__ stats) {
  __shared__ v16bf lds_w[1024];       // 32 KB bf16 C_w, layout [o][k]
  __shared__ float lds_bh[HH];
  __shared__ float lds_agg[HH];
  __shared__ float lds_es[HH];
  __shared__ float lds_es2[HH];

  const int tid  = threadIdx.x;
  const int wave = tid >> 5;
  const int lane = tid & 31;
  const int bi   = blockIdx.x;        // b*V + i
  const int b    = bi >> 8;

  // stage weights into LDS (2048 uint4)
  {
    const uint4* src = (const uint4*)cwb;
    uint4* dst = (uint4*)lds_w;
#pragma unroll
    for (int q = 0; q < 8; ++q) dst[tid + 256 * q] = src[tid + 256 * q];
  }
  if (tid < HH) {
    lds_bh[tid]  = Bh[(size_t)bi * HH + tid];
    lds_agg[tid] = 0.f;
    lds_es[tid]  = 0.f;
    lds_es2[tid] = 0.f;
  }
  __syncthreads();

  float aggl[8], esl[8], es2l[8];
#pragma unroll
  for (int t = 0; t < 8; ++t) { aggl[t] = 0.f; esl[t] = 0.f; es2l[t] = 0.f; }

  const int khalf = (lane & 16) ? 16 : 0;   // A-matrix K half (contiguous, matches B)
  const int koffB = (lane & 16) ? 16 : 0;   // B-matrix K half (contiguous, matches A)

  for (int mt = 0; mt < 2; ++mt) {
    const int jbase  = wave * 32 + mt * 16;           // 16 j-rows per M-tile
    const int jrow_a = jbase + (lane & 15);           // A-matrix row for this lane
    const int rbase  = jbase + ((lane >> 4) << 3);    // C/D-matrix M base for this lane

    // ---- load A fragments (e tile, f32 -> bf16), 4 K-steps ----
    v16bf af[4];
    const float* erow = e + ((size_t)bi * VV + jrow_a) * HH;
#pragma unroll
    for (int s = 0; s < 4; ++s) {
#pragma unroll
      for (int c = 0; c < 2; ++c) {
        const float4 p0 = *(const float4*)(erow + s * 32 + khalf + 8 * c);
        const float4 p1 = *(const float4*)(erow + s * 32 + khalf + 8 * c + 4);
        af[s][8 * c + 0] = (__bf16)p0.x; af[s][8 * c + 1] = (__bf16)p0.y;
        af[s][8 * c + 2] = (__bf16)p0.z; af[s][8 * c + 3] = (__bf16)p0.w;
        af[s][8 * c + 4] = (__bf16)p1.x; af[s][8 * c + 5] = (__bf16)p1.y;
        af[s][8 * c + 6] = (__bf16)p1.z; af[s][8 * c + 7] = (__bf16)p1.w;
      }
    }

    // graph bytes for this lane's 8 output rows (graph==true => non-edge => zero)
    const unsigned long long gbits =
        *(const unsigned long long*)(graph + (size_t)bi * VV + rbase);

#pragma unroll
    for (int t = 0; t < 8; ++t) {                     // 8 N-tiles of 16 channels
      v8f acc = {0.f, 0.f, 0.f, 0.f, 0.f, 0.f, 0.f, 0.f};
#pragma unroll
      for (int s = 0; s < 4; ++s) {                   // K = 4 x 32
        const int widx = ((16 * t + (lane & 15)) * HH + s * 32 + koffB) >> 4;
        v16bf bf = lds_w[widx];
        acc = __builtin_amdgcn_wmma_f32_16x16x32_bf16(
            false, af[s], false, bf, (short)0, acc, false, false);
      }
      const int o = 16 * t + (lane & 15);
      const float bhv = lds_bh[o];
#pragma unroll
      for (int r = 0; r < 8; ++r) {
        const int jr = rbase + r;
        const size_t src = ((size_t)(b * VV + jr)) * HH + o;
        const float ah  = Ahat[src];
        const float vhv = Vh[src];
        const float en  = acc[r] + ah + bhv;
        out_e[((size_t)bi * VV + jr) * HH + o] = en;  // pre-BN e_new
        esl[t]  += en;
        es2l[t] += en * en;
        const float gate = fast_sigmoid(en);
        const bool dead  = ((gbits >> (8 * r)) & 0xffull) != 0ull;
        aggl[t] += dead ? 0.0f : gate * vhv;
      }
    }
  }

  // fold lane partials into LDS per-channel accumulators
#pragma unroll
  for (int t = 0; t < 8; ++t) {
    const int o = 16 * t + (lane & 15);
    atomicAdd(&lds_agg[o], aggl[t]);
    atomicAdd(&lds_es[o],  esl[t]);
    atomicAdd(&lds_es2[o], es2l[t]);
  }
  __syncthreads();

  if (tid < HH) {
    const float hpre = Uh[(size_t)bi * HH + tid] + lds_agg[tid];
    out_h[(size_t)bi * HH + tid] = hpre;              // pre-BN h_new
    atomicAdd(&stats[tid],        lds_es[tid]);
    atomicAdd(&stats[128 + tid],  lds_es2[tid]);
    atomicAdd(&stats[256 + tid],  hpre);
    atomicAdd(&stats[384 + tid],  hpre * hpre);
  }
}

// ---------------------------------------------------------------------------
// K2: fold BN stats into per-channel scale/shift.
// ss layout: [0..127]=a_e [128..255]=b_e [256..383]=a_h [384..511]=b_h
// ---------------------------------------------------------------------------
__global__ void gcn_stats(const float* __restrict__ stats,
                          const float* __restrict__ ge, const float* __restrict__ be,
                          const float* __restrict__ gh, const float* __restrict__ bh,
                          float* __restrict__ ss) {
  const int o = threadIdx.x;
  const float inv_ne = 1.0f / (float)(BB * VV * VV);
  const float me = stats[o] * inv_ne;
  const float ve = stats[128 + o] * inv_ne - me * me;
  const float re = rsqrtf(ve + 1e-5f);
  const float ae = ge[o] * re;
  ss[o]       = ae;
  ss[128 + o] = be[o] - me * ae;

  const float inv_nh = 1.0f / (float)BV;
  const float mh = stats[256 + o] * inv_nh;
  const float vh = stats[384 + o] * inv_nh - mh * mh;
  const float rh = rsqrtf(vh + 1e-5f);
  const float ah = gh[o] * rh;
  ss[256 + o] = ah;
  ss[384 + o] = bh[o] - mh * ah;
}

// ---------------------------------------------------------------------------
// K3/K4: streaming finalize (in place on d_out): out = in + relu(pre*a + b)
// ---------------------------------------------------------------------------
__global__ void gcn_fin_h(float* __restrict__ out_h, const float* __restrict__ h_in,
                          const float* __restrict__ ss) {
  const int i = blockIdx.x * blockDim.x + threadIdx.x;   // float4 index
  const int o = (i * 4) & 127;
  const float4 a = *(const float4*)(ss + 256 + o);
  const float4 c = *(const float4*)(ss + 384 + o);
  float4 p = ((const float4*)out_h)[i];
  const float4 x = ((const float4*)h_in)[i];
  p.x = x.x + fmaxf(0.f, fmaf(p.x, a.x, c.x));
  p.y = x.y + fmaxf(0.f, fmaf(p.y, a.y, c.y));
  p.z = x.z + fmaxf(0.f, fmaf(p.z, a.z, c.z));
  p.w = x.w + fmaxf(0.f, fmaf(p.w, a.w, c.w));
  ((float4*)out_h)[i] = p;
}

__global__ void gcn_fin_e(float* __restrict__ out_e, const float* __restrict__ e_in,
                          const float* __restrict__ ss) {
  const size_t i = (size_t)blockIdx.x * blockDim.x + threadIdx.x;  // float4 index
  const int o = (int)((i * 4) & 127);
  const float4 a = *(const float4*)(ss + o);
  const float4 c = *(const float4*)(ss + 128 + o);
  float4 p = ((const float4*)out_e)[i];
  const float4 x = ((const float4*)e_in)[i];
  p.x = x.x + fmaxf(0.f, fmaf(p.x, a.x, c.x));
  p.y = x.y + fmaxf(0.f, fmaf(p.y, a.y, c.y));
  p.z = x.z + fmaxf(0.f, fmaf(p.z, a.z, c.z));
  p.w = x.w + fmaxf(0.f, fmaf(p.w, a.w, c.w));
  ((float4*)out_e)[i] = p;
}

// ---------------------------------------------------------------------------
extern "C" void kernel_launch(void* const* d_in, const int* in_sizes, int n_in,
                              void* d_out, int out_size, void* d_ws, size_t ws_size,
                              hipStream_t stream) {
  const float* h   = (const float*)d_in[0];
  const float* e   = (const float*)d_in[1];
  const unsigned char* graph = (const unsigned char*)d_in[2];
  const float* U_w = (const float*)d_in[3];
  const float* U_b = (const float*)d_in[4];
  const float* V_w = (const float*)d_in[5];
  const float* V_b = (const float*)d_in[6];
  const float* A_w = (const float*)d_in[7];
  const float* A_b = (const float*)d_in[8];
  const float* B_w = (const float*)d_in[9];
  const float* B_b = (const float*)d_in[10];
  const float* C_w = (const float*)d_in[11];
  const float* C_b = (const float*)d_in[12];
  const float* gamma_h = (const float*)d_in[13];
  const float* beta_h  = (const float*)d_in[14];
  const float* gamma_e = (const float*)d_in[15];
  const float* beta_e  = (const float*)d_in[16];

  char* ws = (char*)d_ws;
  float*  Uh    = (float*)(ws + 0);
  float*  Vh    = (float*)(ws + 1 * 1048576);
  float*  Ahat  = (float*)(ws + 2 * 1048576);
  float*  Bh    = (float*)(ws + 3 * 1048576);
  __bf16* cwb   = (__bf16*)(ws + 4 * 1048576);
  float*  stats = (float*)(ws + 4 * 1048576 + 32768);   // 512 floats
  float*  ss    = stats + 512;                          // 512 floats

  float* out_h = (float*)d_out;
  float* out_e = out_h + BVH;

  gcn_init<<<64, 256, 0, stream>>>(C_w, cwb, stats);
  gcn_linear4<<<BVH / 256, 256, 0, stream>>>(h, U_w, U_b, V_w, V_b, A_w, A_b,
                                             B_w, B_b, C_b, Uh, Vh, Ahat, Bh);
  gcn_main<<<BV, 256, 0, stream>>>(e, graph, Ahat, Vh, Bh, Uh, cwb,
                                   out_e, out_h, stats);
  gcn_stats<<<1, 128, 0, stream>>>(stats, gamma_e, beta_e, gamma_h, beta_h, ss);
  gcn_fin_h<<<BVH / 4 / 256, 256, 0, stream>>>(out_h, h, ss);
  gcn_fin_e<<<(unsigned)(NE_ELEMS / 4 / 256), 256, 0, stream>>>(out_e, e, ss);
}
